// SimpleMPGNN_49349174231248
// MI455X (gfx1250) — matrix operand order repaired and checked by reading, not compiled, not run on hardware
//
#include <hip/hip_runtime.h>
#include <hip/hip_bf16.h>

// Problem constants (from reference)
#define N_NODES 25000
#define N_EDGES 400000
#define F_IN    32
#define F_EDGE  64
#define HID     16
#define NCLS    10
#define NH1     288      // edge-MLP-1 hidden
#define NO1     512      // 32*16
#define NH2     112      // edge-MLP-2 hidden
#define NO2     160      // 16*10

typedef __attribute__((ext_vector_type(16))) _Float16 v16h;
typedef __attribute__((ext_vector_type(8)))  float    v8f;
typedef __attribute__((ext_vector_type(4)))  unsigned v4u;
typedef __attribute__((ext_vector_type(8)))  int      v8i;
typedef __attribute__((ext_vector_type(4)))  int      v4i;

union V16HU { v16h v; unsigned u[8]; };

__device__ __forceinline__ unsigned pkh(float a, float b) {
  union { _Float16 hh[2]; unsigned u; } t;
  t.hh[0] = (_Float16)a; t.hh[1] = (_Float16)b;
  return t.u;
}

__device__ __forceinline__ float uph(unsigned u, int hi) {
  union { unsigned u; _Float16 hh[2]; } t;
  t.u = u;
  return (float)t.hh[hi];
}

__device__ __forceinline__ v8f wmma16(v16h a, v16h b, v8f c) {
  return __builtin_amdgcn_wmma_f32_16x16x32_f16(false, a, false, b, (short)0, c,
                                                false, false);
}

// All packed fragment buffers are stored PRE-SWIZZLED: fragment = 64 16-byte
// chunks, chunk c stored at c ^ ((c>>4)&1) (XOR lane-bit3 into chunk-bit0).
// A wave's two b128 reads then touch all 64 LDS banks exactly twice (the
// minimum for 512B).  The same two-chunk read works from global memory.
__device__ __forceinline__ v16h frag_ld(const _Float16* base, int frag,
                                        int lane) {
  const int b = (lane >> 3) & 1;
  const uint4* p = (const uint4*)(base + (size_t)frag * 512 + lane * 16);
  union { v16h v; uint4 q[2]; } t;
  t.q[0] = p[b];
  t.q[1] = p[b ^ 1];
  return t.v;
}

// ---------------------------------------------------------------------------
// Tensor Data Mover: 1-D DMA of n8 8-byte elements global -> LDS.
// D# built per CDNA5 ISA 8.3/8.4: group0 = {count=1, lds_addr, global_addr,
// type=2}; group1 = {data_size=8B, tensor_dim0=tile_dim0=n8, tensor_dim1=1,
// tile_dim1=1, dim0_stride=n8}.  Issued by one wave; tracked by TENSORcnt.
// This toolchain exposes the 6-arg builtin form (extra int32x8 group + cpol).
// ---------------------------------------------------------------------------
#if __has_builtin(__builtin_amdgcn_tensor_load_to_lds)
#define HAVE_TDM 1
__device__ __forceinline__ void tdm_load_1d(void* lds_dst, const void* gsrc,
                                            unsigned n8) {
  unsigned long long ga = (unsigned long long)(size_t)gsrc;
  v4u g0;
  g0[0] = 1u;                                            // count=1, user mode
  g0[1] = (unsigned)(size_t)lds_dst;                     // LDS byte address
  g0[2] = (unsigned)(ga & 0xFFFFFFFFu);                  // global_addr lo
  g0[3] = (unsigned)((ga >> 32) & 0x1FFFFFFu) | (2u << 30);  // hi | type=2
  v8i g1;
  g1[0] = (int)(3u << 16);                               // data_size = 8B
  g1[1] = (int)((n8 & 0xFFFFu) << 16);                   // tensor_dim0 lo
  g1[2] = (int)((n8 >> 16) | (1u << 16));                // dim0 hi | dim1=1
  g1[3] = (int)((n8 & 0xFFFFu) << 16);                   // tile_dim0 = n8
  g1[4] = 1;                                             // tile_dim1 = 1
  g1[5] = (int)n8;                                       // dim0_stride lo
  g1[6] = 0;
  g1[7] = 0;
  v4i z4 = {0, 0, 0, 0};
  v8i z8 = {0, 0, 0, 0, 0, 0, 0, 0};
  __builtin_amdgcn_tensor_load_to_lds(g0, g1, z4, z4, z8, 0);
}
#else
#define HAVE_TDM 0
#endif

// LDS fill: TDM DMA when available (wave 0 issues, waits TENSORcnt), else a
// plain cooperative copy (buffers are already swizzled in global memory).
__device__ __forceinline__ void fill_lds(void* lds_dst, const void* gsrc,
                                         int bytes) {
#if HAVE_TDM
  if (threadIdx.x < 32) {
    tdm_load_1d(lds_dst, gsrc, (unsigned)(bytes >> 3));
    __builtin_amdgcn_s_wait_tensorcnt(0);
  }
#else
  uint4* s4 = (uint4*)lds_dst;
  const uint4* g4 = (const uint4*)gsrc;
  for (int i = threadIdx.x; i < (bytes >> 4); i += 256) s4[i] = g4[i];
#endif
}

// Build GEMM2 B-fragment from two H^T C-tiles via half-wave swap:
// lane<16 keeps tile 2kt2 rows 0..7, lane>=16 keeps tile 2kt2+1 rows 8..15;
// the other 8 values come from the partner lane (xor 16).
__device__ __forceinline__ v16h repack_bfrag(const float* ca, const float* cb,
                                             int h) {
  unsigned ku[4], su[4];
#pragma unroll
  for (int j = 0; j < 4; ++j) {
    ku[j] = h ? pkh(cb[2 * j], cb[2 * j + 1]) : pkh(ca[2 * j], ca[2 * j + 1]);
    su[j] = h ? pkh(ca[2 * j], ca[2 * j + 1]) : pkh(cb[2 * j], cb[2 * j + 1]);
  }
  V16HU f;
#pragma unroll
  for (int j = 0; j < 4; ++j) {
    unsigned rv = __shfl_xor(su[j], 16, 32);
    f.u[j]     = h ? rv    : ku[j];
    f.u[4 + j] = h ? ku[j] : rv;
  }
  return f.v;
}

// swizzled output index for packed fragments
__device__ __forceinline__ int pack_idx(int frag, int l, int j) {
  int c = (l << 1) | (j >> 3);
  int cs = c ^ ((c >> 4) & 1);
  return (frag << 9) + (cs << 3) + (j & 7);
}

// ---------------------------------------------------------------------------
// Weight repack: W[K,Ncols] fp32 -> A-fragment layout (A = W^T, 16x32 f16
// tiles), stored pre-swizzled.
// ---------------------------------------------------------------------------
__global__ void pack_at(const float* __restrict__ W, _Float16* __restrict__ out,
                        int K, int Ncols, int Tt, int KT) {
  int id = blockIdx.x * blockDim.x + threadIdx.x;
  int total = Tt * KT * 512;
  if (id >= total) return;
  int j = id & 15;
  int l = (id >> 4) & 31;
  int frag = id >> 9;
  int kt = frag % KT;
  int t  = frag / KT;
  int col = t * 16 + (l & 15);
  int h = l >> 4;
  int k = kt * 32 + h * 16 + j;
  float v = 0.0f;
  if (col < Ncols && k < K) v = W[(size_t)k * Ncols + col];
  out[pack_idx(frag, l, j)] = (_Float16)v;
}

// w2b [112,160] -> padded A-frags: 16 col-tiles (cols 16i+o <- 10i+o, o<10),
// K padded 112 -> 128 (4 k-tiles), pad entries = 0.  Pre-swizzled.
__global__ void pack_w2b(const float* __restrict__ W, _Float16* __restrict__ out) {
  int id = blockIdx.x * blockDim.x + threadIdx.x;
  if (id >= 16 * 4 * 512) return;
  int j = id & 15;
  int l = (id >> 4) & 31;
  int frag = id >> 9;
  int kt = frag & 3;
  int i  = frag >> 2;        // = n2 = input channel
  int o  = l & 15;
  int h = l >> 4;
  int k = kt * 32 + h * 16 + j;
  float v = 0.0f;
  if (o < 10 && k < NH2) v = W[(size_t)k * NO2 + i * 10 + o];
  out[pack_idx(frag, l, j)] = (_Float16)v;
}

__global__ void zero_f32(float* __restrict__ p, int n) {
  int i = blockIdx.x * blockDim.x + threadIdx.x;
  if (i < n) p[i] = 0.0f;
}

// ---------------------------------------------------------------------------
// Layer-1 edge pipeline: one wave = 32 edges (two 16-edge groups sharing the
// LDS-resident A-fragment stream; W2a^T = 288 KB in 320 KB/WGP LDS).
// ---------------------------------------------------------------------------
__global__ __launch_bounds__(256, 1) void edge_layer1(
    const float* __restrict__ x, const int* __restrict__ src,
    const int* __restrict__ dst, const float* __restrict__ ea,
    const _Float16* __restrict__ pa1, const _Float16* __restrict__ pa2,
    const float* __restrict__ b1a, const float* __restrict__ b2a,
    float* __restrict__ agg1) {
  extern __shared__ char smraw[];
  _Float16* sm = (_Float16*)smraw;            // 288 KB of W2a^T fragments

  fill_lds(smraw, pa2, 32 * 9 * 512 * 2);
  __syncthreads();

  const int lane = threadIdx.x & 31;
  const int m = lane & 15;
  const int h = lane >> 4;
  const int wid = blockIdx.x * 8 + (threadIdx.x >> 5);
  const int nw  = gridDim.x * 8;
  const int NT  = N_EDGES / 32;

  for (int tile = wid; tile < NT; tile += nw) {
    if (tile + nw < NT)
      __builtin_prefetch(ea + (size_t)(tile + nw) * 32 * F_EDGE, 0, 1);

    const int e0 = tile * 32 + m;
    const int e1 = e0 + 16;
    const int sv0 = src[e0], dv0 = dst[e0];
    const int sv1 = src[e1], dv1 = dst[e1];

    // source-node feature rows, packed f16 pairs (16 u32 per group)
    unsigned xpA[F_IN / 2], xpB[F_IN / 2];
    {
      const float4* pA = (const float4*)(x + (size_t)sv0 * F_IN);
      const float4* pB = (const float4*)(x + (size_t)sv1 * F_IN);
#pragma unroll
      for (int q = 0; q < 8; ++q) {
        float4 a = pA[q], b = pB[q];
        xpA[2 * q] = pkh(a.x, a.y); xpA[2 * q + 1] = pkh(a.z, a.w);
        xpB[2 * q] = pkh(b.x, b.y); xpB[2 * q + 1] = pkh(b.z, b.w);
      }
    }

    // GEMM1: groups run sequentially to cap register pressure
    v16h bhA[9], bhB[9];
#pragma unroll
    for (int g = 0; g < 2; ++g) {
      const int eg = g ? e1 : e0;
      v16h bea[2];
#pragma unroll
      for (int kt = 0; kt < 2; ++kt) {
        const float* p = ea + (size_t)eg * F_EDGE + kt * 32 + h * 16;
#pragma unroll
        for (int j = 0; j < 16; ++j) bea[kt][j] = (_Float16)p[j];
      }
#pragma unroll
      for (int kt2 = 0; kt2 < 9; ++kt2) {
        float ca[8], cb[8];
#pragma unroll
        for (int hf = 0; hf < 2; ++hf) {
          const int t1 = 2 * kt2 + hf;
          v16h a0 = frag_ld(pa1, t1 * 2 + 0, lane);
          v16h a1 = frag_ld(pa1, t1 * 2 + 1, lane);
          v8f acc = {};
          acc = wmma16(a0, bea[0], acc);
          acc = wmma16(a1, bea[1], acc);
          const float* bb = b1a + t1 * 16 + 8 * h;
          float* c = hf ? cb : ca;
#pragma unroll
          for (int r = 0; r < 8; ++r) c[r] = fmaxf(acc[r] + bb[r], 0.0f);
        }
        v16h f = repack_bfrag(ca, cb, h);
        if (g) bhB[kt2] = f; else bhA[kt2] = f;
      }
    }

    // GEMM2: A-fragments streamed just-in-time from LDS, each feeds 2 WMMAs
    float msgA[8] = {0, 0, 0, 0, 0, 0, 0, 0};
    float msgB[8] = {0, 0, 0, 0, 0, 0, 0, 0};
#pragma unroll
    for (int n2 = 0; n2 < 32; ++n2) {
      v8f accA = {}, accB = {};
#pragma unroll
      for (int kt = 0; kt < 9; ++kt) {
        v16h a = frag_ld(sm, n2 * 9 + kt, lane);
        accA = wmma16(a, bhA[kt], accA);
        accB = wmma16(a, bhB[kt], accB);
      }
      const float* bb = b2a + n2 * 16 + 8 * h;   // col = 16*n2 + 8h + r
      const float xvA = uph(xpA[n2 >> 1], n2 & 1);
      const float xvB = uph(xpB[n2 >> 1], n2 & 1);
#pragma unroll
      for (int r = 0; r < 8; ++r) {
        msgA[r] += xvA * (accA[r] + bb[r]);
        msgB[r] += xvB * (accB[r] + bb[r]);
      }
    }
#pragma unroll
    for (int r = 0; r < 8; ++r) {
      unsafeAtomicAdd(&agg1[(size_t)dv0 * HID + 8 * h + r], msgA[r]);
      unsafeAtomicAdd(&agg1[(size_t)dv1 * HID + 8 * h + r], msgB[r]);
    }
  }
}

// ---------------------------------------------------------------------------
// Layer-2 edge pipeline (dims padded: K2 112->128, cols 160->256 via 16i+o).
// Both weight pools staged in LDS (78 KB) via one TDM DMA (contiguous in ws).
// ---------------------------------------------------------------------------
__global__ __launch_bounds__(256, 1) void edge_layer2(
    const float* __restrict__ x1, const int* __restrict__ src,
    const int* __restrict__ dst, const float* __restrict__ ea,
    const _Float16* __restrict__ pa1b, const _Float16* __restrict__ pa2b,
    const float* __restrict__ b1b, const float* __restrict__ b2b,
    float* __restrict__ agg2) {
  extern __shared__ char smraw[];
  _Float16* sm1 = (_Float16*)smraw;            // 7*2*512 halves  (W1b^T)
  _Float16* sm2 = sm1 + 7 * 2 * 512;           // 16*4*512 halves (W2b^T padded)
  fill_lds(smraw, pa1b, (7 * 2 * 512 + 16 * 4 * 512) * 2);
  __syncthreads();

  const int lane = threadIdx.x & 31;
  const int m = lane & 15;
  const int h = lane >> 4;
  const int wid = blockIdx.x * 8 + (threadIdx.x >> 5);
  const int nw  = gridDim.x * 8;
  const int NT  = N_EDGES / 32;

  for (int tile = wid; tile < NT; tile += nw) {
    if (tile + nw < NT)
      __builtin_prefetch(ea + (size_t)(tile + nw) * 32 * F_EDGE, 0, 1);

    const int e0 = tile * 32 + m;
    const int e1 = e0 + 16;
    const int sv0 = src[e0], dv0 = dst[e0];
    const int sv1 = src[e1], dv1 = dst[e1];

    unsigned xpA[HID / 2], xpB[HID / 2];
    {
      const float4* pA = (const float4*)(x1 + (size_t)sv0 * HID);
      const float4* pB = (const float4*)(x1 + (size_t)sv1 * HID);
#pragma unroll
      for (int q = 0; q < 4; ++q) {
        float4 a = pA[q], b = pB[q];
        xpA[2 * q] = pkh(a.x, a.y); xpA[2 * q + 1] = pkh(a.z, a.w);
        xpB[2 * q] = pkh(b.x, b.y); xpB[2 * q + 1] = pkh(b.z, b.w);
      }
    }

    v16h bhA[4], bhB[4];
#pragma unroll
    for (int g = 0; g < 2; ++g) {
      const int eg = g ? e1 : e0;
      v16h bea[2];
#pragma unroll
      for (int kt = 0; kt < 2; ++kt) {
        const float* p = ea + (size_t)eg * F_EDGE + kt * 32 + h * 16;
#pragma unroll
        for (int j = 0; j < 16; ++j) bea[kt][j] = (_Float16)p[j];
      }
#pragma unroll
      for (int kt2 = 0; kt2 < 4; ++kt2) {
        float ca[8], cb[8];
        {
          const int t1 = 2 * kt2;
          v16h a0 = frag_ld(sm1, t1 * 2 + 0, lane);
          v16h a1 = frag_ld(sm1, t1 * 2 + 1, lane);
          v8f acc = {};
          acc = wmma16(a0, bea[0], acc);
          acc = wmma16(a1, bea[1], acc);
          const float* bb = b1b + t1 * 16 + 8 * h;
#pragma unroll
          for (int r = 0; r < 8; ++r) ca[r] = fmaxf(acc[r] + bb[r], 0.0f);
        }
        const int t1b = 2 * kt2 + 1;
        if (t1b < 7) {
          v16h a0 = frag_ld(sm1, t1b * 2 + 0, lane);
          v16h a1 = frag_ld(sm1, t1b * 2 + 1, lane);
          v8f acc = {};
          acc = wmma16(a0, bea[0], acc);
          acc = wmma16(a1, bea[1], acc);
          const float* bb = b1b + t1b * 16 + 8 * h;
#pragma unroll
          for (int r = 0; r < 8; ++r) cb[r] = fmaxf(acc[r] + bb[r], 0.0f);
        } else {
#pragma unroll
          for (int r = 0; r < 8; ++r) { cb[r] = 0.0f; }   // K padding 112..127
        }
        v16h f = repack_bfrag(ca, cb, h);
        if (g) bhB[kt2] = f; else bhA[kt2] = f;
      }
    }

    float msgA[8] = {0, 0, 0, 0, 0, 0, 0, 0};
    float msgB[8] = {0, 0, 0, 0, 0, 0, 0, 0};
#pragma unroll
    for (int n2 = 0; n2 < 16; ++n2) {
      v8f accA = {}, accB = {};
#pragma unroll
      for (int kt = 0; kt < 4; ++kt) {
        v16h a = frag_ld(sm2, n2 * 4 + kt, lane);
        accA = wmma16(a, bhA[kt], accA);
        accB = wmma16(a, bhB[kt], accB);
      }
      const float xvA = uph(xpA[n2 >> 1], n2 & 1);
      const float xvB = uph(xpB[n2 >> 1], n2 & 1);
#pragma unroll
      for (int r = 0; r < 8; ++r) {
        int o = 8 * h + r;
        float bb = (o < 10) ? b2b[n2 * 10 + o] : 0.0f;
        msgA[r] += xvA * (accA[r] + bb);
        msgB[r] += xvB * (accB[r] + bb);
      }
    }
#pragma unroll
    for (int r = 0; r < 8; ++r) {
      int o = 8 * h + r;
      if (o < 10) {
        unsafeAtomicAdd(&agg2[(size_t)dv0 * NCLS + o], msgA[r]);
        unsafeAtomicAdd(&agg2[(size_t)dv1 * NCLS + o], msgB[r]);
      }
    }
  }
}

// x1 = relu(x @ root1 + agg1 + bias1)
__global__ void node_layer1(const float* __restrict__ x,
                            const float* __restrict__ agg1,
                            const float* __restrict__ root1,
                            const float* __restrict__ bias1,
                            float* __restrict__ x1) {
  int t = blockIdx.x * blockDim.x + threadIdx.x;
  if (t >= N_NODES * HID) return;
  int v = t >> 4, o = t & 15;
  float s = bias1[o] + agg1[t];
  const float* xr = x + (size_t)v * F_IN;
#pragma unroll
  for (int i = 0; i < F_IN; ++i) s += xr[i] * root1[i * HID + o];
  x1[t] = fmaxf(s, 0.0f);
}

// out = log_softmax(x1 @ root2 + agg2 + bias2)
__global__ void node_layer2(const float* __restrict__ x1,
                            const float* __restrict__ agg2,
                            const float* __restrict__ root2,
                            const float* __restrict__ bias2,
                            float* __restrict__ out) {
  int v = blockIdx.x * blockDim.x + threadIdx.x;
  if (v >= N_NODES) return;
  float xr[HID];
#pragma unroll
  for (int i = 0; i < HID; ++i) xr[i] = x1[(size_t)v * HID + i];
  float lg[NCLS];
  float mx = -1e30f;
#pragma unroll
  for (int c = 0; c < NCLS; ++c) {
    float s = bias2[c] + agg2[(size_t)v * NCLS + c];
#pragma unroll
    for (int i = 0; i < HID; ++i) s += xr[i] * root2[i * NCLS + c];
    lg[c] = s;
    mx = fmaxf(mx, s);
  }
  float se = 0.0f;
#pragma unroll
  for (int c = 0; c < NCLS; ++c) se += expf(lg[c] - mx);
  float lse = logf(se);
#pragma unroll
  for (int c = 0; c < NCLS; ++c) out[(size_t)v * NCLS + c] = lg[c] - mx - lse;
}

// ---------------------------------------------------------------------------
extern "C" void kernel_launch(void* const* d_in, const int* in_sizes, int n_in,
                              void* d_out, int out_size, void* d_ws,
                              size_t ws_size, hipStream_t stream) {
  const float* x    = (const float*)d_in[0];
  const int*   ei   = (const int*)d_in[1];
  const float* ea   = (const float*)d_in[2];
  const float* w1a  = (const float*)d_in[3];
  const float* b1a  = (const float*)d_in[4];
  const float* w2a  = (const float*)d_in[5];
  const float* b2a  = (const float*)d_in[6];
  const float* root1= (const float*)d_in[7];
  const float* bias1= (const float*)d_in[8];
  const float* w1b  = (const float*)d_in[9];
  const float* b1b  = (const float*)d_in[10];
  const float* w2b  = (const float*)d_in[11];
  const float* b2b  = (const float*)d_in[12];
  const float* root2= (const float*)d_in[13];
  const float* bias2= (const float*)d_in[14];
  const int* src = ei;
  const int* dst = ei + N_EDGES;

  char* ws = (char*)d_ws;
  float* agg1 = (float*)(ws);                         // N*16 f32 = 1.6 MB
  float* x1   = (float*)(ws + 1600000);               // N*16 f32
  float* agg2 = (float*)(ws + 3200000);               // N*10 f32 = 1.0 MB
  _Float16* pa1  = (_Float16*)(ws + 4200448);         // 18*2*512 halves
  _Float16* pa2  = pa1  + 18 * 2 * 512;               // 32*9*512
  _Float16* pa1b = pa2  + 32 * 9 * 512;               // 7*2*512
  _Float16* pa2b = pa1b + 7 * 2 * 512;                // 16*4*512  (~4.6 MB total)

  hipLaunchKernelGGL(zero_f32, dim3((N_NODES * HID + 255) / 256), dim3(256), 0,
                     stream, agg1, N_NODES * HID);
  hipLaunchKernelGGL(zero_f32, dim3((N_NODES * NCLS + 255) / 256), dim3(256), 0,
                     stream, agg2, N_NODES * NCLS);

  hipLaunchKernelGGL(pack_at, dim3((18 * 2 * 512 + 255) / 256), dim3(256), 0,
                     stream, w1a, pa1, F_EDGE, NH1, 18, 2);
  hipLaunchKernelGGL(pack_at, dim3((32 * 9 * 512 + 255) / 256), dim3(256), 0,
                     stream, w2a, pa2, NH1, NO1, 32, 9);
  hipLaunchKernelGGL(pack_at, dim3((7 * 2 * 512 + 255) / 256), dim3(256), 0,
                     stream, w1b, pa1b, F_EDGE, NH2, 7, 2);
  hipLaunchKernelGGL(pack_w2b, dim3((16 * 4 * 512 + 255) / 256), dim3(256), 0,
                     stream, w2b, pa2b);

  // Layer 1: 288 KB dynamic LDS (swizzled W2a^T fragments), persistent blocks
  hipLaunchKernelGGL(edge_layer1, dim3(512), dim3(256), 32 * 9 * 512 * 2,
                     stream, x, src, dst, ea, pa1, pa2, b1a, b2a, agg1);
  hipLaunchKernelGGL(node_layer1, dim3((N_NODES * HID + 255) / 256), dim3(256),
                     0, stream, x, agg1, root1, bias1, x1);
  // Layer 2: 78 KB dynamic LDS (swizzled W1b^T + padded W2b^T fragments)
  hipLaunchKernelGGL(edge_layer2, dim3(512), dim3(256),
                     (7 * 2 * 512 + 16 * 4 * 512) * 2, stream, x1, src, dst, ea,
                     pa1b, pa2b, b1b, b2b, agg2);
  hipLaunchKernelGGL(node_layer2, dim3((N_NODES + 255) / 256), dim3(256), 0,
                     stream, x1, agg2, root2, bias2, (float*)d_out);
}
